// FastFraudGNN_1657857376906
// MI455X (gfx1250) — matrix operand order repaired and checked by reading, not compiled, not run on hardware
//
#include <hip/hip_runtime.h>
#include <hip/hip_bf16.h>
#include <math.h>

#define IN_C 128
#define HID  128

typedef __attribute__((ext_vector_type(16))) _Float16 v16h;
typedef __attribute__((ext_vector_type(8)))  _Float16 v8h;
typedef __attribute__((ext_vector_type(4)))  _Float16 v4h;
typedef __attribute__((ext_vector_type(8)))  float    v8f;

// ---------------- zero fill (float4 granularity) ----------------
__global__ void k_zero_f4(float4* __restrict__ p, long n4) {
  long i = (long)blockIdx.x * blockDim.x + threadIdx.x;
  long stride = (long)gridDim.x * blockDim.x;
  float4 z = make_float4(0.f, 0.f, 0.f, 0.f);
  for (; i < n4; i += stride) p[i] = z;
}

// ---------------- degree accumulation (f32 hardware atomics) ----------------
__global__ void k_degree(const int* __restrict__ dst, float* __restrict__ deg, int E) {
  int i = blockIdx.x * blockDim.x + threadIdx.x;
  int stride = gridDim.x * blockDim.x;
  for (; i < E; i += stride)
    __hip_atomic_fetch_add(&deg[dst[i]], 1.0f, __ATOMIC_RELAXED,
                           __HIP_MEMORY_SCOPE_AGENT);
}

// deg -> dinv in place: rsqrt(deg + 1)  (+1 = self loop; always > 0)
__global__ void k_dinv(float* __restrict__ deg, int Nn) {
  int i = blockIdx.x * blockDim.x + threadIdx.x;
  if (i < Nn) deg[i] = rsqrtf(deg[i] + 1.0f);
}

// ---------------- row-major f32 -> row-major f16 (4 elems / thread) ---------
__global__ void k_cvt_f16(const float* __restrict__ in, _Float16* __restrict__ out,
                          long n4) {
  long i = (long)blockIdx.x * blockDim.x + threadIdx.x;
  long stride = (long)gridDim.x * blockDim.x;
  for (; i < n4; i += stride) {
    float4 v = ((const float4*)in)[i];
    v4h h;
    h[0] = (_Float16)v.x; h[1] = (_Float16)v.y;
    h[2] = (_Float16)v.z; h[3] = (_Float16)v.w;
    ((v4h*)out)[i] = h;
  }
}

// ---------------- pack W[128,128] f32 into WMMA B-fragment order, f16 -------
// Layout: Bp[((ntile*4 + ktile)*32 + lane)*16 + j]
//   = W[ktile*32 + (j>=8 ? 16 : 0) + (lane>=16 ? 8 : 0) + (j&7)][ntile*16 + (lane&15)]
__global__ void k_pack_w(const float* __restrict__ W, _Float16* __restrict__ Bp) {
  int idx = blockIdx.x * blockDim.x + threadIdx.x;   // 0..16383
  int j    = idx & 15;
  int lane = (idx >> 4) & 31;
  int kt   = (idx >> 9) & 3;
  int nt   = idx >> 11;
  int k = kt * 32 + ((j >> 3) << 4) + ((lane >> 4) << 3) + (j & 7);
  int n = nt * 16 + (lane & 15);
  Bp[idx] = (_Float16)W[k * HID + n];
}

// ---------------- C = A[M,128](f16) @ Bp(packed f16) -> f32 -----------------
// block: 256 threads = 8 waves. blockIdx.x = 16-row tile, wave = 16-col tile.
// Hot loop: 4 x {2x b128 A-load + 2x b128 B-load + v_wmma_f32_16x16x32_f16}
__global__ void k_gemm_wmma(const _Float16* __restrict__ A,
                            const _Float16* __restrict__ Bp,
                            float* __restrict__ C, int M) {
  const int lane = threadIdx.x & 31;
  const int wave = threadIdx.x >> 5;          // 0..7 -> output column tile
  const int l15  = lane & 15;
  const int hi   = lane >> 4;                 // half-wave select
  const int rowBase = blockIdx.x * 16;
  const int nBase   = wave * 16;

  const _Float16* arow  = A + (size_t)(rowBase + l15) * IN_C + hi * 8;
  const _Float16* bfrag = Bp + (((size_t)wave * 4) * 32 + lane) * 16;

  v8f acc = {};
#pragma unroll
  for (int kt = 0; kt < 4; ++kt) {
    // A fragment (16x32 f16): lane row = l15; halves {ka..ka+7, ka+16..ka+23}
    v8h alo = *(const v8h*)(arow + kt * 32);
    v8h ahi = *(const v8h*)(arow + kt * 32 + 16);
    v16h a, b;
#pragma unroll
    for (int j = 0; j < 8; ++j) { a[j] = alo[j]; a[j + 8] = ahi[j]; }
    // B fragment: contiguous 32B per lane, pre-packed
    b = *(const v16h*)(bfrag + (size_t)kt * 32 * 16);
    acc = __builtin_amdgcn_wmma_f32_16x16x32_f16(false, a, false, b,
                                                 (short)0, acc, false, false);
  }
  // D layout: VGPR r -> M = r (lanes 0-15) / r+8 (lanes 16-31), N = l15.
  float* crow = C + (size_t)rowBase * HID + nBase + l15;
#pragma unroll
  for (int r = 0; r < 8; ++r)
    crow[(size_t)(r + hi * 8) * HID] = acc[r];
}

// ---------------- edge scatter-add: one wave per edge, float4 per lane ------
__global__ void k_agg(const float* __restrict__ t, const int* __restrict__ src,
                      const int* __restrict__ dst, const float* __restrict__ dinv,
                      float* __restrict__ agg, int E) {
  int gid  = blockIdx.x * blockDim.x + threadIdx.x;
  int lane = gid & 31;
  int wid  = gid >> 5;
  int nw   = (gridDim.x * blockDim.x) >> 5;
  for (int e = wid; e < E; e += nw) {
    int s = src[e];
    int d = dst[e];
    int e2 = e + nw;
    if (e2 < E)   // hide next random-gather latency (global_prefetch_b8)
      __builtin_prefetch(t + (size_t)src[e2] * HID + 4 * lane, 0, 0);
    float w = dinv[s] * dinv[d];
    float4 v = *(const float4*)(t + (size_t)s * HID + 4 * lane);
    float* ad = agg + (size_t)d * HID + 4 * lane;
    __hip_atomic_fetch_add(ad + 0, w * v.x, __ATOMIC_RELAXED, __HIP_MEMORY_SCOPE_AGENT);
    __hip_atomic_fetch_add(ad + 1, w * v.y, __ATOMIC_RELAXED, __HIP_MEMORY_SCOPE_AGENT);
    __hip_atomic_fetch_add(ad + 2, w * v.z, __ATOMIC_RELAXED, __HIP_MEMORY_SCOPE_AGENT);
    __hip_atomic_fetch_add(ad + 3, w * v.w, __ATOMIC_RELAXED, __HIP_MEMORY_SCOPE_AGENT);
  }
}

// ---------------- relu(agg + dinv^2*t + bias) -> f16 (feeds next GEMM) ------
__global__ void k_finalize_f16(const float* __restrict__ agg, const float* __restrict__ t,
                               const float* __restrict__ dinv, const float* __restrict__ bias,
                               _Float16* __restrict__ out, int Nn) {
  int gid  = blockIdx.x * blockDim.x + threadIdx.x;
  int lane = gid & 31;
  int i    = gid >> 5;
  if (i >= Nn) return;
  float di = dinv[i];
  float w  = di * di;                      // self-loop norm
  float4 a  = *(const float4*)(agg  + (size_t)i * HID + 4 * lane);
  float4 tv = *(const float4*)(t    + (size_t)i * HID + 4 * lane);
  float4 b4 = *(const float4*)(bias + 4 * lane);
  v4h r;
  r[0] = (_Float16)fmaxf(a.x + w * tv.x + b4.x, 0.f);
  r[1] = (_Float16)fmaxf(a.y + w * tv.y + b4.y, 0.f);
  r[2] = (_Float16)fmaxf(a.z + w * tv.z + b4.z, 0.f);
  r[3] = (_Float16)fmaxf(a.w + w * tv.w + b4.w, 0.f);
  *(v4h*)(out + (size_t)i * HID + 4 * lane) = r;
}

// ---------------- relu(agg + dinv^2*t + bias) -> f32 (feeds head) -----------
__global__ void k_finalize_f32(const float* __restrict__ agg, const float* __restrict__ t,
                               const float* __restrict__ dinv, const float* __restrict__ bias,
                               float* __restrict__ out, int Nn) {
  int gid  = blockIdx.x * blockDim.x + threadIdx.x;
  int lane = gid & 31;
  int i    = gid >> 5;
  if (i >= Nn) return;
  float di = dinv[i];
  float w  = di * di;
  float4 a  = *(const float4*)(agg  + (size_t)i * HID + 4 * lane);
  float4 tv = *(const float4*)(t    + (size_t)i * HID + 4 * lane);
  float4 b4 = *(const float4*)(bias + 4 * lane);
  float4 r;
  r.x = fmaxf(a.x + w * tv.x + b4.x, 0.f);
  r.y = fmaxf(a.y + w * tv.y + b4.y, 0.f);
  r.z = fmaxf(a.z + w * tv.z + b4.z, 0.f);
  r.w = fmaxf(a.w + w * tv.w + b4.w, 0.f);
  *(float4*)(out + (size_t)i * HID + 4 * lane) = r;
}

// ---------------- head: logits = h @ Wfc + bfc, log_softmax -----------------
__global__ void k_head(const float* __restrict__ h, const float* __restrict__ Wfc,
                       const float* __restrict__ bfc, float* __restrict__ out, int Nn) {
  int gid  = blockIdx.x * blockDim.x + threadIdx.x;
  int lane = gid & 31;
  int i    = gid >> 5;
  if (i >= Nn) return;
  float4 v = *(const float4*)(h + (size_t)i * HID + 4 * lane);
  float hv[4] = {v.x, v.y, v.z, v.w};
  float z0 = 0.f, z1 = 0.f;
#pragma unroll
  for (int j = 0; j < 4; ++j) {
    int c = 4 * lane + j;
    z0 += hv[j] * Wfc[c * 2 + 0];
    z1 += hv[j] * Wfc[c * 2 + 1];
  }
#pragma unroll
  for (int off = 16; off >= 1; off >>= 1) {
    z0 += __shfl_xor(z0, off, 32);
    z1 += __shfl_xor(z1, off, 32);
  }
  if (lane == 0) {
    z0 += bfc[0];
    z1 += bfc[1];
    float m   = fmaxf(z0, z1);
    float lse = m + logf(expf(z0 - m) + expf(z1 - m));
    out[2 * (size_t)i + 0] = z0 - lse;
    out[2 * (size_t)i + 1] = z1 - lse;
  }
}

extern "C" void kernel_launch(void* const* d_in, const int* in_sizes, int n_in,
                              void* d_out, int out_size, void* d_ws, size_t ws_size,
                              hipStream_t stream) {
  const float* x   = (const float*)d_in[0];
  const int*   ei  = (const int*)d_in[1];   // [2, E] row-major int32
  const float* W1  = (const float*)d_in[2];
  const float* b1  = (const float*)d_in[3];
  const float* W2  = (const float*)d_in[4];
  const float* b2  = (const float*)d_in[5];
  const float* Wfc = (const float*)d_in[6];
  const float* bfc = (const float*)d_in[7];

  const int Nn = in_sizes[0] / IN_C;        // 100000
  const int E  = in_sizes[1] / 2;           // 1600000
  const int* src = ei;
  const int* dst = ei + E;

  // workspace layout: dinv | bufA f32 | bufB f32 | bufH f16 | Wp1 | Wp2
  char* ws = (char*)d_ws;
  size_t offDinv = 0;
  size_t szDinv  = (((size_t)Nn * sizeof(float)) + 255) & ~(size_t)255;
  size_t szFeat  = (((size_t)Nn * HID * sizeof(float)) + 255) & ~(size_t)255;
  size_t szFeatH = (((size_t)Nn * HID * sizeof(_Float16)) + 255) & ~(size_t)255;
  size_t szWp    = ((size_t)HID * HID * sizeof(_Float16) + 255) & ~(size_t)255;

  float*    dinv = (float*)(ws + offDinv);
  float*    bufA = (float*)(ws + szDinv);
  float*    bufB = (float*)(ws + szDinv + szFeat);
  _Float16* bufH = (_Float16*)(ws + szDinv + 2 * szFeat);
  _Float16* Wp1  = (_Float16*)(ws + szDinv + 2 * szFeat + szFeatH);
  _Float16* Wp2  = (_Float16*)(ws + szDinv + 2 * szFeat + szFeatH + szWp);

  const long n4feat = (long)Nn * HID / 4;
  const int  blkN   = (Nn * 32 + 255) / 256;     // 1 wave / node kernels

  // degrees -> dinv
  k_zero_f4<<<256, 256, 0, stream>>>((float4*)dinv, (long)(Nn + 3) / 4);
  k_degree<<<1024, 256, 0, stream>>>(dst, dinv, E);
  k_dinv<<<(Nn + 255) / 256, 256, 0, stream>>>(dinv, Nn);

  // one-time (per launch) f16 prep: pack weights, convert x
  k_pack_w<<<64, 256, 0, stream>>>(W1, Wp1);
  k_pack_w<<<64, 256, 0, stream>>>(W2, Wp2);
  k_cvt_f16<<<2048, 256, 0, stream>>>(x, bufH, n4feat);

  // layer 1: t0 = x @ W1 ; agg ; h1 = relu(agg + dinv^2*t0 + b1) -> f16
  k_gemm_wmma<<<Nn / 16, 256, 0, stream>>>(bufH, Wp1, bufA, Nn);
  k_zero_f4<<<2048, 256, 0, stream>>>((float4*)bufB, n4feat);
  k_agg<<<4096, 256, 0, stream>>>(bufA, src, dst, dinv, bufB, E);
  k_finalize_f16<<<blkN, 256, 0, stream>>>(bufB, bufA, dinv, b1, bufH, Nn);

  // layer 2: t1 = h1 @ W2 ; agg ; h2 = relu(agg + dinv^2*t1 + b2) -> f32
  k_gemm_wmma<<<Nn / 16, 256, 0, stream>>>(bufH, Wp2, bufA, Nn);
  k_zero_f4<<<2048, 256, 0, stream>>>((float4*)bufB, n4feat);
  k_agg<<<4096, 256, 0, stream>>>(bufA, src, dst, dinv, bufB, E);
  k_finalize_f32<<<blkN, 256, 0, stream>>>(bufB, bufA, dinv, b2, bufB, Nn);

  // classifier head + log_softmax
  k_head<<<blkN, 256, 0, stream>>>(bufB, Wfc, bfc, (float*)d_out, Nn);
}
